// GNNEncoder_87814901334539
// MI455X (gfx1250) — matrix operand order repaired and checked by reading, not compiled, not run on hardware
//
#include <hip/hip_runtime.h>

typedef __attribute__((ext_vector_type(16))) _Float16 v16h;
typedef __attribute__((ext_vector_type(8)))  _Float16 v8h;
typedef __attribute__((ext_vector_type(8)))  float    v8f;

// ---------------- degree / normalization ----------------

__global__ void deg_init_kernel(float* __restrict__ deg, int n) {
    int i = blockIdx.x * blockDim.x + threadIdx.x;
    if (i < n) deg[i] = 1.0f;   // self loop contributes 1 to every node
}

__global__ void deg_accum_kernel(const int* __restrict__ dst, float* __restrict__ deg, int E) {
    int e = blockIdx.x * blockDim.x + threadIdx.x;
    if (e < E) atomicAdd(&deg[dst[e]], 1.0f);
}

__global__ void dinv_kernel(float* __restrict__ deg, int n) {
    int i = blockIdx.x * blockDim.x + threadIdx.x;
    if (i < n) {
        float d = deg[i];
        deg[i] = (d > 0.0f) ? rsqrtf(d) : 0.0f;   // in-place: deg -> dinv
    }
}

// ---------------- operand packing (fp32 -> fp16, K zero-padded to Kp) ----------------

__global__ void pack_a_f16_kernel(const float* __restrict__ A, _Float16* __restrict__ Ap,
                                  int nrows, int K, int Kp) {
    long i = blockIdx.x * (long)blockDim.x + threadIdx.x;
    long total = (long)nrows * Kp;
    if (i >= total) return;
    int r = (int)(i / Kp);
    int k = (int)(i - (long)r * Kp);
    Ap[i] = (k < K) ? (_Float16)A[(size_t)r * K + k] : (_Float16)0.0f;
}

// W[K x Cout] -> Wt[Cout x Kp] (transposed, f16, zero-padded)
__global__ void pack_wt_f16_kernel(const float* __restrict__ W, _Float16* __restrict__ Wt,
                                   int K, int Kp, int Cout) {
    int i = blockIdx.x * blockDim.x + threadIdx.x;
    int total = Cout * Kp;
    if (i >= total) return;
    int n = i / Kp;
    int k = i - n * Kp;
    Wt[i] = (k < K) ? (_Float16)W[(size_t)k * Cout + n] : (_Float16)0.0f;
}

// ---------------- WMMA GEMM: H[nrows x Cout] = Ap[nrows x Kp] @ Wt^T ----------------
// One wave computes a 16x64 output strip: 4 col-tiles share one A fragment.
// Fragments use contiguous k = 16*(lane/16) + s in both A and B (not the ISA table order); all loads contiguous & unpredicated.

__global__ void gemm_wmma_kernel(const _Float16* __restrict__ Ap, const _Float16* __restrict__ Wt,
                                 float* __restrict__ H, int nrows, int Kp, int Cout) {
    int wave = (int)((blockIdx.x * blockDim.x + threadIdx.x) >> 5);
    int lane = threadIdx.x & 31;
    int grpPerRow = Cout >> 6;                       // 64-wide column groups
    int total = (nrows >> 4) * grpPerRow;
    if (wave >= total) return;                       // uniform per wave: EXEC all-1s inside

    int mt = wave / grpPerRow;
    int g  = wave - mt * grpPerRow;
    int m0 = mt << 4;
    int n0 = g << 6;
    int hid = lane >> 4;                             // half-wave id
    int mr  = lane & 15;

    const _Float16* arow = Ap + (size_t)(m0 + mr) * Kp;
    const _Float16* wrow0 = Wt + (size_t)(n0 + mr) * Kp + (hid << 4);
    size_t wstride16 = (size_t)16 * Kp;              // advance 16 output columns

    v8f c0 = {}, c1 = {}, c2 = {}, c3 = {};
    for (int kc = 0; kc < Kp; kc += 32) {
        // A fragment: two contiguous 8-half runs (16B each)
        v8h alo = *(const v8h*)(arow + kc + (hid << 4));
        v8h ahi = *(const v8h*)(arow + kc + 8 + (hid << 4));
        v16h a = __builtin_shufflevector(alo, ahi, 0,1,2,3,4,5,6,7,8,9,10,11,12,13,14,15);
        // B fragments: one contiguous 32B run per col-tile
        v16h b0 = *(const v16h*)(wrow0 + kc);
        v16h b1 = *(const v16h*)(wrow0 + wstride16 + kc);
        v16h b2 = *(const v16h*)(wrow0 + 2 * wstride16 + kc);
        v16h b3 = *(const v16h*)(wrow0 + 3 * wstride16 + kc);
        c0 = __builtin_amdgcn_wmma_f32_16x16x32_f16(false, a, false, b0, (short)0, c0, false, false);
        c1 = __builtin_amdgcn_wmma_f32_16x16x32_f16(false, a, false, b1, (short)0, c1, false, false);
        c2 = __builtin_amdgcn_wmma_f32_16x16x32_f16(false, a, false, b2, (short)0, c2, false, false);
        c3 = __builtin_amdgcn_wmma_f32_16x16x32_f16(false, a, false, b3, (short)0, c3, false, false);
    }
#pragma unroll
    for (int r = 0; r < 8; ++r) {
        float* orow = H + (size_t)(m0 + (hid << 3) + r) * Cout + mr;
        orow[n0 +  0] = c0[r];
        orow[n0 + 16] = c1[r];
        orow[n0 + 32] = c2[r];
        orow[n0 + 48] = c3[r];
    }
}

// ---------------- aggregation ----------------

template <int C>
__global__ void bias_init_kernel(float* __restrict__ out, const float* __restrict__ b, long total) {
    long i = blockIdx.x * (long)blockDim.x + threadIdx.x;
    if (i < total) out[i] = b[(int)(i & (C - 1))];
}

// one thread per (edge-or-selfloop, 4-channel group); gather h[src], scale, scatter-add
template <int C4>
__global__ void aggregate_kernel(const float4* __restrict__ h, const int* __restrict__ src,
                                 const int* __restrict__ dst, const float* __restrict__ dinv,
                                 float* __restrict__ out, int E, int n) {
    long tid = blockIdx.x * (long)blockDim.x + threadIdx.x;
    long total = (long)(E + n) * C4;
    if (tid >= total) return;
    int e  = (int)(tid >> __builtin_ctz(C4));        // C4 is a power of two
    int c4 = (int)(tid & (C4 - 1));
    int s, d;
    if (e < E) { s = src[e]; d = dst[e]; }
    else       { s = e - E; d = s; }                 // self loop
    float nm = dinv[s] * dinv[d];
    float4 hv = h[(long)s * C4 + c4];
    float* o = out + ((long)d * C4 + c4) * 4;
    atomicAdd(o + 0, hv.x * nm);
    atomicAdd(o + 1, hv.y * nm);
    atomicAdd(o + 2, hv.z * nm);
    atomicAdd(o + 3, hv.w * nm);
}

__global__ void relu_kernel(float* __restrict__ x, long total) {
    long i = blockIdx.x * (long)blockDim.x + threadIdx.x;
    if (i < total) x[i] = fmaxf(x[i], 0.0f);
}

// ---------------- orchestration ----------------

static inline unsigned cdiv_l(long a, long b) { return (unsigned)((a + b - 1) / b); }

extern "C" void kernel_launch(void* const* d_in, const int* in_sizes, int n_in,
                              void* d_out, int out_size, void* d_ws, size_t ws_size,
                              hipStream_t stream) {
    const int IN = 13, H1 = 128, H2 = 64, OUT = 64;
    const float* x   = (const float*)d_in[0];
    const int*   ei  = (const int*)d_in[1];
    const float* W1  = (const float*)d_in[2];
    const float* b1  = (const float*)d_in[3];
    const float* W2  = (const float*)d_in[4];
    const float* b2  = (const float*)d_in[5];
    const float* W3  = (const float*)d_in[6];
    const float* b3  = (const float*)d_in[7];

    const int N = in_sizes[0] / IN;
    const int E = in_sizes[1] / 2;
    const int* src = ei;          // edge_index[0]
    const int* dst = ei + E;      // edge_index[1]

    // workspace layout (64B aligned slices):
    //   bufA | bufB : N x 128 fp32 each   (GEMM out / aggregation out, ping-pong)
    //   dinv        : N fp32
    //   Ap          : N x 128 f16 (packed GEMM A operand)
    //   Wt          : 128*128 f16 (packed transposed weights)
    char* p = (char*)d_ws;
    auto align64 = [](size_t v) { return (v + 63) & ~(size_t)63; };
    size_t off = 0;
    float* bufA = (float*)(p + off); off = align64(off + (size_t)N * 128 * sizeof(float));
    float* bufB = (float*)(p + off); off = align64(off + (size_t)N * 128 * sizeof(float));
    float* dinv = (float*)(p + off); off = align64(off + (size_t)N * sizeof(float));
    _Float16* Aph = (_Float16*)(p + off); off = align64(off + (size_t)N * 128 * sizeof(_Float16));
    _Float16* Wth = (_Float16*)(p + off);

    const int TB = 256;

    // --- degrees -> dinv ---
    deg_init_kernel<<<cdiv_l(N, TB), TB, 0, stream>>>(dinv, N);
    deg_accum_kernel<<<cdiv_l(E, TB), TB, 0, stream>>>(dst, dinv, E);
    dinv_kernel<<<cdiv_l(N, TB), TB, 0, stream>>>(dinv, N);

    // --- layer 1: 13 -> 128 (Kp = 32) ---
    {
        const int K = IN, Kp = 32, C = H1;
        pack_a_f16_kernel<<<cdiv_l((long)N * Kp, TB), TB, 0, stream>>>(x, Aph, N, K, Kp);
        pack_wt_f16_kernel<<<cdiv_l(C * Kp, TB), TB, 0, stream>>>(W1, Wth, K, Kp, C);
        int waves = (N / 16) * (C / 64);
        gemm_wmma_kernel<<<cdiv_l(waves, 8), TB, 0, stream>>>(Aph, Wth, bufA, N, Kp, C);
        long total = (long)N * C;
        bias_init_kernel<128><<<cdiv_l(total, TB), TB, 0, stream>>>(bufB, b1, total);
        long work = (long)(E + N) * (C / 4);
        aggregate_kernel<32><<<cdiv_l(work, TB), TB, 0, stream>>>((const float4*)bufA, src, dst,
                                                                  dinv, bufB, E, N);
        relu_kernel<<<cdiv_l(total, TB), TB, 0, stream>>>(bufB, total);
    }

    // --- layer 2: 128 -> 64 (Kp = 128) ---
    {
        const int K = H1, Kp = 128, C = H2;
        pack_a_f16_kernel<<<cdiv_l((long)N * Kp, TB), TB, 0, stream>>>(bufB, Aph, N, K, Kp);
        pack_wt_f16_kernel<<<cdiv_l(C * Kp, TB), TB, 0, stream>>>(W2, Wth, K, Kp, C);
        int waves = (N / 16) * (C / 64);
        gemm_wmma_kernel<<<cdiv_l(waves, 8), TB, 0, stream>>>(Aph, Wth, bufA, N, Kp, C);
        long total = (long)N * C;
        bias_init_kernel<64><<<cdiv_l(total, TB), TB, 0, stream>>>(bufB, b2, total);
        long work = (long)(E + N) * (C / 4);
        aggregate_kernel<16><<<cdiv_l(work, TB), TB, 0, stream>>>((const float4*)bufA, src, dst,
                                                                  dinv, bufB, E, N);
        relu_kernel<<<cdiv_l(total, TB), TB, 0, stream>>>(bufB, total);
    }

    // --- layer 3: 64 -> 64 (Kp = 64, no relu, write d_out) ---
    {
        const int K = H2, Kp = 64, C = OUT;
        pack_a_f16_kernel<<<cdiv_l((long)N * Kp, TB), TB, 0, stream>>>(bufB, Aph, N, K, Kp);
        pack_wt_f16_kernel<<<cdiv_l(C * Kp, TB), TB, 0, stream>>>(W3, Wth, K, Kp, C);
        int waves = (N / 16) * (C / 64);
        gemm_wmma_kernel<<<cdiv_l(waves, 8), TB, 0, stream>>>(Aph, Wth, bufA, N, Kp, C);
        long total = (long)N * C;
        float* out = (float*)d_out;
        bias_init_kernel<64><<<cdiv_l(total, TB), TB, 0, stream>>>(out, b3, total);
        long work = (long)(E + N) * (C / 4);
        aggregate_kernel<16><<<cdiv_l(work, TB), TB, 0, stream>>>((const float4*)bufA, src, dst,
                                                                  dinv, out, E, N);
    }
}
